// SwinBlock_11879879544557
// MI455X (gfx1250) — matrix operand order repaired and checked by reading, not compiled, not run on hardware
//
#include <hip/hip_runtime.h>
#include <cstdint>
#include <cstddef>

// ---------------------------------------------------------------------------
// Types for gfx1250 WMMA (wave32): v_wmma_f32_16x16x32_bf16
// ---------------------------------------------------------------------------
typedef __bf16 bf16;
typedef __attribute__((ext_vector_type(16))) __bf16 v16bf;
typedef __attribute__((ext_vector_type(8)))  __bf16 bf16x8;
typedef __attribute__((ext_vector_type(8)))  float  v8f;
typedef __attribute__((ext_vector_type(4)))  int    v4i;

#define TOKENS   262144      // 4 * 256 * 256
#define CDIM     180
#define KPAD     192         // 180 padded to 6*32
#define HEADS    6
#define HDIM     30
#define HPAD     32
#define NQKV     540
#define NQKVPAD  544
#define HIDDEN   720
#define HIDPAD   736         // 720 padded to 23*32
#define ATT_SCALE 0.18257418583505536f  // 1/sqrt(30)

// ---------------------------------------------------------------------------
// CDNA5 async global->LDS copy (ASYNCcnt-tracked), with safe fallback
// ---------------------------------------------------------------------------
#if defined(__AMDGCN__) && __has_builtin(__builtin_amdgcn_global_load_async_to_lds_b128) && \
    __has_builtin(__builtin_amdgcn_s_wait_asynccnt)
#define HAVE_ASYNC_LDS 1
#else
#define HAVE_ASYNC_LDS 0
#endif

typedef __attribute__((address_space(1))) v4i gv4i;   // global int4
typedef __attribute__((address_space(3))) v4i lv4i;   // LDS int4

__device__ __forceinline__ void cp16_g2lds(void* lds_dst, const void* gsrc) {
#if HAVE_ASYNC_LDS
  __builtin_amdgcn_global_load_async_to_lds_b128(
      (gv4i*)(void*)gsrc,   // generic -> addrspace(1) v4i*
      (lv4i*)lds_dst,       // generic -> addrspace(3) v4i*
      0, 0);
#else
  *reinterpret_cast<bf16x8*>(lds_dst) = *reinterpret_cast<const bf16x8*>(gsrc);
#endif
}

__device__ __forceinline__ void async_copies_done() {
#if HAVE_ASYNC_LDS
  __builtin_amdgcn_s_wait_asynccnt(0);
#endif
}

// Prefetch a bf16 weight matrix into cache (global_prefetch_b8), cooperatively.
__device__ __forceinline__ void prefetch_weights(const bf16* w, int elems, int tid) {
  for (int i = tid * 64; i < elems; i += 256 * 64)   // 128B cacheline stride
    __builtin_prefetch(w + i, 0, 3);
}

__device__ __forceinline__ v8f wmma_bf16(v16bf a, v16bf b, v8f c) {
  return __builtin_amdgcn_wmma_f32_16x16x32_bf16(false, a, false, b, (short)0, c,
                                                 false, false);
}

// A fragment: row-major src [M][ld] bf16.  lane: m = l&15, k = (l>>4)*8,
// elements 0..7 <- K=k..k+7, elements 8..15 <- K=k+16..k+23   (ISA 7.12.2)
__device__ __forceinline__ v16bf load_frag_a(const bf16* src, int ld,
                                             int mBase, int kBase, int lane) {
  int m = mBase + (lane & 15);
  int k = kBase + ((lane >> 4) << 3);
  bf16x8 lo = *reinterpret_cast<const bf16x8*>(src + (size_t)m * ld + k);
  bf16x8 hi = *reinterpret_cast<const bf16x8*>(src + (size_t)m * ld + k + 16);
  v16bf r;
#pragma unroll
  for (int i = 0; i < 8; ++i) { r[i] = lo[i]; r[i + 8] = hi[i]; }
  return r;
}

// B fragment from TRANSPOSED weight [N][ld(K)]: lane: n = l&15, k = (l>>4)*8,
// elements 0..7 <- K=k..k+7, elements 8..15 <- K=k+16..k+23 (matches A).
__device__ __forceinline__ v16bf load_frag_b(const bf16* srcT, int ld,
                                             int nBase, int kBase, int lane) {
  int n = nBase + (lane & 15);
  int k = kBase + ((lane >> 4) << 3);
  bf16x8 lo = *reinterpret_cast<const bf16x8*>(srcT + (size_t)n * ld + k);
  bf16x8 hi = *reinterpret_cast<const bf16x8*>(srcT + (size_t)n * ld + k + 16);
  v16bf r;
#pragma unroll
  for (int i = 0; i < 8; ++i) { r[i] = lo[i]; r[i + 8] = hi[i]; }
  return r;
}

// ---------------------------------------------------------------------------
// Kernel 1: weight convert fp32 [K][N] -> bf16 transposed padded [Npad][Kpad]
// ---------------------------------------------------------------------------
__global__ void convw_kernel(const float* __restrict__ src, bf16* __restrict__ dstT,
                             int K, int N, int Kpad, int Npad) {
  int idx = blockIdx.x * blockDim.x + threadIdx.x;
  int total = Kpad * Npad;
  if (idx >= total) return;
  int n = idx / Kpad;
  int k = idx - n * Kpad;
  float v = (k < K && n < N) ? src[(size_t)k * N + n] : 0.0f;
  dstT[idx] = (bf16)v;
}

// ---------------------------------------------------------------------------
// Kernel 2/5: LayerNorm (one wave32 per token) -> bf16, stride KPAD, zero pad
// ---------------------------------------------------------------------------
__global__ __launch_bounds__(256) void ln_kernel(const float* __restrict__ x,
                                                 const float* __restrict__ scale,
                                                 const float* __restrict__ bias,
                                                 bf16* __restrict__ out) {
  int gw = (blockIdx.x * 256 + threadIdx.x) >> 5;   // global wave = token
  int lane = threadIdx.x & 31;
  if (gw >= TOKENS) return;
  const float* row = x + (size_t)gw * CDIM;
  float s = 0.f, s2 = 0.f;
  for (int c = lane; c < CDIM; c += 32) { float v = row[c]; s += v; s2 += v * v; }
#pragma unroll
  for (int off = 16; off; off >>= 1) {
    s  += __shfl_xor(s, off, 32);
    s2 += __shfl_xor(s2, off, 32);
  }
  float mean = s * (1.0f / CDIM);
  float var  = s2 * (1.0f / CDIM) - mean * mean;
  float inv  = rsqrtf(var + 1e-5f);
  bf16* orow = out + (size_t)gw * KPAD;
  for (int c = lane; c < KPAD; c += 32) {
    float v = (c < CDIM) ? ((row[c] - mean) * inv * scale[c] + bias[c]) : 0.0f;
    orow[c] = (bf16)v;
  }
}

// ---------------------------------------------------------------------------
// Kernel 3: fused window attention (QKV GEMM + softmax(QK^T)V), 1 WG / window
// ---------------------------------------------------------------------------
__device__ __forceinline__ int win_token(int b, int wy, int wx, int r) {
  return b * 65536 + (wy * 8 + (r >> 3)) * 256 + wx * 8 + (r & 7);
}

__global__ __launch_bounds__(256) void attn_kernel(const bf16* __restrict__ hbuf,
                                                   const bf16* __restrict__ qkvwT,
                                                   const float* __restrict__ qkv_b,
                                                   bf16* __restrict__ obuf) {
  __shared__ __align__(16) bf16 win[64][KPAD];        // 24.6 KB
  __shared__ __align__(16) bf16 qs[HEADS][64][HPAD];  // 24.6 KB
  __shared__ __align__(16) bf16 ks[HEADS][64][HPAD];  // 24.6 KB
  __shared__ __align__(16) bf16 vT[HEADS][HPAD][64];  // 24.6 KB
  __shared__ __align__(16) float sc[64][64];          // 16.4 KB
  __shared__ __align__(16) bf16  pm[64][64];          //  8.2 KB

  int widx = blockIdx.x;
  int b  = widx >> 10;
  int wy = (widx & 1023) >> 5;
  int wx = widx & 31;
  int tid = threadIdx.x;
  int wave = tid >> 5, lane = tid & 31;

  // async-gather 64 window tokens into LDS (4 threads per row, 16B chunks)
  {
    int r = tid >> 2, part = tid & 3;
    int tok = win_token(b, wy, wx, r);
    const bf16* src = hbuf + (size_t)tok * KPAD;
    for (int i = part; i < KPAD / 8; i += 4)
      cp16_g2lds(&win[r][i * 8], src + i * 8);
  }
  prefetch_weights(qkvwT, NQKVPAD * KPAD, tid);
  // zero head-dim padding (d = 30,31) and obuf pad columns
  for (int i = tid; i < HEADS * 64 * 2; i += 256) {
    int h = i >> 7, rem = i & 127, m = rem >> 1, d = 30 + (rem & 1);
    qs[h][m][d] = (bf16)0.f;
    ks[h][m][d] = (bf16)0.f;
    vT[h][d][m] = (bf16)0.f;
  }
  for (int i = tid; i < 64 * (KPAD - CDIM); i += 256) {
    int r = i / (KPAD - CDIM), c = CDIM + i % (KPAD - CDIM);
    obuf[(size_t)win_token(b, wy, wx, r) * KPAD + c] = (bf16)0.f;
  }
  async_copies_done();
  __syncthreads();

  // ---- QKV GEMM: [64 x 192] @ [192 x 544] -> q/k/v in LDS ----
  // per-wave: mt fixed (A fragments loop-invariant, held in VGPRs), nt streams
  for (int t = wave; t < 4 * (NQKVPAD / 16); t += 8) {
    int mt = t & 3, nt = t >> 2;
    v8f acc = {};
#pragma unroll
    for (int kt = 0; kt < KPAD / 32; ++kt) {
      v16bf a  = load_frag_a(&win[0][0], KPAD, mt * 16, kt * 32, lane);
      v16bf bw = load_frag_b(qkvwT, KPAD, nt * 16, kt * 32, lane);
      acc = wmma_bf16(a, bw, acc);
    }
    int n = nt * 16 + (lane & 15);
    if (n < NQKV) {
      int seg = n / CDIM, n180 = n % CDIM;
      int h = n180 / HDIM, d = n180 % HDIM;
      float bv = qkv_b[n];
#pragma unroll
      for (int v = 0; v < 8; ++v) {
        int m = mt * 16 + ((lane >> 4) << 3) + v;
        float val = acc[v] + bv;
        if      (seg == 0) qs[h][m][d] = (bf16)val;
        else if (seg == 1) ks[h][m][d] = (bf16)val;
        else               vT[h][d][m] = (bf16)val;
      }
    }
  }
  __syncthreads();

  // ---- per-head attention ----
  for (int h = 0; h < HEADS; ++h) {
    // scores = Q @ K^T  (16 tiles of 16x16, K=32)
    for (int t = wave; t < 16; t += 8) {
      int mt = t >> 2, nt = t & 3;
      v16bf a  = load_frag_a(&qs[h][0][0], HPAD, mt * 16, 0, lane);
      v16bf bk = load_frag_b(&ks[h][0][0], HPAD, nt * 16, 0, lane);
      v8f acc = {};
      acc = wmma_bf16(a, bk, acc);
      int n = nt * 16 + (lane & 15);
#pragma unroll
      for (int v = 0; v < 8; ++v)
        sc[mt * 16 + ((lane >> 4) << 3) + v][n] = acc[v];
    }
    __syncthreads();
    // softmax rows (fp32), 8 rows per wave, 2 cols per lane
    for (int r = wave; r < 64; r += 8) {
      float v0 = sc[r][lane]      * ATT_SCALE;
      float v1 = sc[r][lane + 32] * ATT_SCALE;
      float mx = fmaxf(v0, v1);
#pragma unroll
      for (int off = 16; off; off >>= 1) mx = fmaxf(mx, __shfl_xor(mx, off, 32));
      float e0 = expf(v0 - mx), e1 = expf(v1 - mx);
      float sm = e0 + e1;
#pragma unroll
      for (int off = 16; off; off >>= 1) sm += __shfl_xor(sm, off, 32);
      float inv = 1.0f / sm;
      pm[r][lane]      = (bf16)(e0 * inv);
      pm[r][lane + 32] = (bf16)(e1 * inv);
    }
    __syncthreads();
    // O = P @ V  (4 mtiles x 2 ntiles, K=64)
    for (int t = wave; t < 8; t += 8) {
      int mt = t >> 1, nt = t & 1;
      v8f acc = {};
#pragma unroll
      for (int kt = 0; kt < 2; ++kt) {
        v16bf a  = load_frag_a(&pm[0][0], 64, mt * 16, kt * 32, lane);
        v16bf bv = load_frag_b(&vT[h][0][0], 64, nt * 16, kt * 32, lane);
        acc = wmma_bf16(a, bv, acc);
      }
      int d = nt * 16 + (lane & 15);
      if (d < HDIM) {
        int col = h * HDIM + d;
#pragma unroll
        for (int v = 0; v < 8; ++v) {
          int r = mt * 16 + ((lane >> 4) << 3) + v;
          obuf[(size_t)win_token(b, wy, wx, r) * KPAD + col] = (bf16)acc[v];
        }
      }
    }
    __syncthreads();
  }
}

// ---------------------------------------------------------------------------
// Kernel 4: proj GEMM + bias + residual  ->  d_out (fp32, = x2)
// ---------------------------------------------------------------------------
__global__ __launch_bounds__(256) void proj_kernel(const bf16* __restrict__ obuf,
                                                   const bf16* __restrict__ projwT,
                                                   const float* __restrict__ bias,
                                                   const float* __restrict__ xin,
                                                   float* __restrict__ out) {
  __shared__ __align__(16) bf16 at[64][KPAD];
  int mBlock = blockIdx.x * 64;
  int tid = threadIdx.x, wave = tid >> 5, lane = tid & 31;
  {
    int r = tid >> 2, part = tid & 3;
    const bf16* src = obuf + (size_t)(mBlock + r) * KPAD;
    for (int i = part; i < KPAD / 8; i += 4)
      cp16_g2lds(&at[r][i * 8], src + i * 8);
  }
  prefetch_weights(projwT, KPAD * KPAD, tid);
  async_copies_done();
  __syncthreads();
  for (int t = wave; t < 48; t += 8) {          // 4 mtiles x 12 ntiles
    int mt = t & 3, nt = t >> 2;
    v8f acc = {};
#pragma unroll
    for (int kt = 0; kt < KPAD / 32; ++kt) {
      v16bf a  = load_frag_a(&at[0][0], KPAD, mt * 16, kt * 32, lane);
      v16bf bw = load_frag_b(projwT, KPAD, nt * 16, kt * 32, lane);
      acc = wmma_bf16(a, bw, acc);
    }
    int col = nt * 16 + (lane & 15);
    if (col < CDIM) {
      float bv = bias[col];
#pragma unroll
      for (int v = 0; v < 8; ++v) {
        size_t row = (size_t)(mBlock + mt * 16 + ((lane >> 4) << 3) + v);
        out[row * CDIM + col] = acc[v] + bv + xin[row * CDIM + col];
      }
    }
  }
}

// ---------------------------------------------------------------------------
// Kernel 6: fused MLP — hidden (64x736 bf16) stays in LDS; out += mlp(h2)
// ---------------------------------------------------------------------------
__global__ __launch_bounds__(256) void mlp_kernel(const bf16* __restrict__ hbuf,
                                                  const bf16* __restrict__ w1T,
                                                  const float* __restrict__ b1,
                                                  const bf16* __restrict__ w2T,
                                                  const float* __restrict__ b2,
                                                  float* __restrict__ out) {
  __shared__ __align__(16) bf16 at[64][KPAD];     // 24.6 KB
  __shared__ __align__(16) bf16 hid[64][HIDPAD];  // 94.2 KB
  int mBlock = blockIdx.x * 64;
  int tid = threadIdx.x, wave = tid >> 5, lane = tid & 31;
  {
    int r = tid >> 2, part = tid & 3;
    const bf16* src = hbuf + (size_t)(mBlock + r) * KPAD;
    for (int i = part; i < KPAD / 8; i += 4)
      cp16_g2lds(&at[r][i * 8], src + i * 8);
  }
  prefetch_weights(w1T, HIDDEN * KPAD, tid);
  prefetch_weights(w2T, KPAD * HIDPAD, tid);
  for (int i = tid; i < 64 * (HIDPAD - HIDDEN); i += 256)   // zero hidden K-pad
    hid[i >> 4][HIDDEN + (i & 15)] = (bf16)0.f;
  async_copies_done();
  __syncthreads();
  // phase 1: hidden = gelu(h2 @ w1 + b1)   (4 x 45 tiles)
  for (int t = wave; t < 4 * (HIDDEN / 16); t += 8) {
    int mt = t & 3, nt = t >> 2;
    v8f acc = {};
#pragma unroll
    for (int kt = 0; kt < KPAD / 32; ++kt) {
      v16bf a  = load_frag_a(&at[0][0], KPAD, mt * 16, kt * 32, lane);
      v16bf bw = load_frag_b(w1T, KPAD, nt * 16, kt * 32, lane);
      acc = wmma_bf16(a, bw, acc);
    }
    int n = nt * 16 + (lane & 15);
    float bv = b1[n];
#pragma unroll
    for (int v = 0; v < 8; ++v) {
      int m = mt * 16 + ((lane >> 4) << 3) + v;
      float x = acc[v] + bv;
      float g = 0.5f * x * (1.0f + erff(x * 0.70710678118654752f)); // exact GELU
      hid[m][n] = (bf16)g;
    }
  }
  __syncthreads();
  // phase 2: out += hidden @ w2 + b2   (4 x 12 tiles, K = 736 = 23*32)
  for (int t = wave; t < 48; t += 8) {
    int mt = t & 3, nt = t >> 2;
    v8f acc = {};
#pragma unroll
    for (int kt = 0; kt < HIDPAD / 32; ++kt) {
      v16bf a  = load_frag_a(&hid[0][0], HIDPAD, mt * 16, kt * 32, lane);
      v16bf bw = load_frag_b(w2T, HIDPAD, nt * 16, kt * 32, lane);
      acc = wmma_bf16(a, bw, acc);
    }
    int col = nt * 16 + (lane & 15);
    if (col < CDIM) {
      float bv = b2[col];
#pragma unroll
      for (int v = 0; v < 8; ++v) {
        size_t row = (size_t)(mBlock + mt * 16 + ((lane >> 4) << 3) + v);
        out[row * CDIM + col] += acc[v] + bv;
      }
    }
  }
}

// ---------------------------------------------------------------------------
// Host launcher
// ---------------------------------------------------------------------------
extern "C" void kernel_launch(void* const* d_in, const int* in_sizes, int n_in,
                              void* d_out, int out_size, void* d_ws, size_t ws_size,
                              hipStream_t stream) {
  const float* x      = (const float*)d_in[0];
  const float* n1s    = (const float*)d_in[1];
  const float* n1b    = (const float*)d_in[2];
  const float* qkv_w  = (const float*)d_in[3];
  const float* qkv_b  = (const float*)d_in[4];
  const float* proj_w = (const float*)d_in[5];
  const float* proj_b = (const float*)d_in[6];
  const float* n2s    = (const float*)d_in[7];
  const float* n2b    = (const float*)d_in[8];
  const float* mlp_w1 = (const float*)d_in[9];
  const float* mlp_b1 = (const float*)d_in[10];
  const float* mlp_w2 = (const float*)d_in[11];
  const float* mlp_b2 = (const float*)d_in[12];
  float* out = (float*)d_out;

  char* ws = (char*)d_ws;
  const size_t hbuf_b   = (size_t)TOKENS * KPAD * sizeof(bf16);   // 100.7 MB
  const size_t obuf_b   = hbuf_b;
  const size_t qkvwT_b  = (size_t)NQKVPAD * KPAD * sizeof(bf16);
  const size_t projwT_b = (size_t)KPAD * KPAD * sizeof(bf16);
  const size_t w1T_b    = (size_t)HIDDEN * KPAD * sizeof(bf16);
  bf16* hbuf   = (bf16*)(ws);
  bf16* obuf   = (bf16*)(ws + hbuf_b);
  bf16* qkvwT  = (bf16*)(ws + hbuf_b + obuf_b);
  bf16* projwT = (bf16*)(ws + hbuf_b + obuf_b + qkvwT_b);
  bf16* w1T    = (bf16*)(ws + hbuf_b + obuf_b + qkvwT_b + projwT_b);
  bf16* w2T    = (bf16*)(ws + hbuf_b + obuf_b + qkvwT_b + projwT_b + w1T_b);

  // weight conversion (fp32 -> transposed padded bf16)
  {
    int t;
    t = NQKVPAD * KPAD; convw_kernel<<<(t + 255) / 256, 256, 0, stream>>>(qkv_w,  qkvwT,  CDIM,   NQKV,   KPAD,   NQKVPAD);
    t = KPAD * KPAD;    convw_kernel<<<(t + 255) / 256, 256, 0, stream>>>(proj_w, projwT, CDIM,   CDIM,   KPAD,   KPAD);
    t = HIDDEN * KPAD;  convw_kernel<<<(t + 255) / 256, 256, 0, stream>>>(mlp_w1, w1T,    CDIM,   HIDDEN, KPAD,   HIDDEN);
    t = KPAD * HIDPAD;  convw_kernel<<<(t + 255) / 256, 256, 0, stream>>>(mlp_w2, w2T,    HIDDEN, CDIM,   HIDPAD, KPAD);
  }

  const int lnBlocks = TOKENS / 8;  // 8 waves (tokens) per 256-thread block
  ln_kernel<<<lnBlocks, 256, 0, stream>>>(x, n1s, n1b, hbuf);
  attn_kernel<<<4096, 256, 0, stream>>>(hbuf, qkvwT, qkv_b, obuf);
  proj_kernel<<<TOKENS / 64, 256, 0, stream>>>(obuf, projwT, proj_b, x, out);
  ln_kernel<<<lnBlocks, 256, 0, stream>>>(out, n2s, n2b, hbuf);   // reuse hbuf
  mlp_kernel<<<TOKENS / 64, 256, 0, stream>>>(hbuf, w1T, mlp_b1, w2T, mlp_b2, out);
}
